// CausalLinearAttention_87943750352898
// MI455X (gfx1250) — matrix operand order repaired and checked by reading, not compiled, not run on hardware
//
#include <hip/hip_runtime.h>
#include <hip/hip_bf16.h>
#include <stdint.h>

#define BATCH 4
#define TLEN 2048
#define D 128
#define DV 128
#define CHUNK 64
#define NC (TLEN / CHUNK)   // 32
#define EPSV 1e-6f

typedef __attribute__((ext_vector_type(16))) __bf16 bf16x16;
typedef __attribute__((ext_vector_type(8)))  float  f32x8;

// ---- fragment gathers from LDS (row-major source, leading dim ld) ----
// A tile 16x32 (MxK) at (row0,k0): lane holds row M=row0+(lane&15);
// element j -> K = k0 + (lane>>4)*16 + j  (contiguous, matches frag_b/frag_bT)
__device__ __forceinline__ bf16x16 frag_a(const __bf16* s, int row0, int k0, int ld, int lane) {
  const int m = row0 + (lane & 15);
  const int h8 = (lane >> 4) << 4;
  bf16x16 a;
#pragma unroll
  for (int j = 0; j < 16; ++j) {
    int k = k0 + j + h8;
    a[j] = s[m * ld + k];
  }
  return a;
}
// B tile 32x16 (KxN) at (k0,col0): lane holds col N=col0+(lane&15);
// half-wave selects K block of 16; element j -> K = kb + j
__device__ __forceinline__ bf16x16 frag_b(const __bf16* s, int k0, int col0, int ld, int lane) {
  const int n = col0 + (lane & 15);
  const int kb = k0 + ((lane >> 4) << 4);
  bf16x16 b;
#pragma unroll
  for (int j = 0; j < 16; ++j) b[j] = s[(kb + j) * ld + n];
  return b;
}
// logical B[k][n] = s[n][k] (transposed source: W^T or K^T) -- contiguous in j
__device__ __forceinline__ bf16x16 frag_bT(const __bf16* s, int k0, int col0, int ld, int lane) {
  const int n = col0 + (lane & 15);
  const int kb = k0 + ((lane >> 4) << 4);
  bf16x16 b;
#pragma unroll
  for (int j = 0; j < 16; ++j) b[j] = s[n * ld + (kb + j)];
  return b;
}

// phi(x) = ELU(x)+1 == (x>0 ? x+1 : exp(x))
__device__ __forceinline__ float phi_act(float x) { return x > 0.f ? x + 1.f : __expf(x); }

// ---- CDNA5 async memory->LDS copy (ASYNCcnt path) ----
__device__ __forceinline__ void async_copy_b128(uint32_t lds_off, const void* gbase, uint32_t goff) {
  asm volatile("global_load_async_to_lds_b128 %0, %1, %2 offset:0"
               :: "v"(lds_off), "v"(goff), "s"(gbase) : "memory");
}
__device__ __forceinline__ void wait_async0() {
  asm volatile("s_wait_asynccnt 0x0" ::: "memory");
}

// =====================================================================
// Pass A: per (batch, chunk): Qf=phi(qW^T+b), Kf=phi(kW^T+b) (bf16 out),
//         chunkKV = Kf^T @ V  [D x DV],  chunkZ = column sums of Kf [D]
// =====================================================================
__global__ __launch_bounds__(256) void cla_passA(
    const float* __restrict__ q, const float* __restrict__ k, const float* __restrict__ v,
    const float* __restrict__ W, const float* __restrict__ bias,
    __bf16* __restrict__ Qf, __bf16* __restrict__ Kf,
    float* __restrict__ ckv, float* __restrict__ cz) {
  const int bc = blockIdx.x;                 // 0 .. BATCH*NC-1
  const int b = bc / NC, c = bc % NC;
  const int tid = threadIdx.x;
  const int lane = tid & 31, wave = tid >> 5;

  __shared__ __bf16 sW[D * D];        // W row-major [e][d]          (32 KB)
  __shared__ __bf16 sX[CHUNK * D];    // input chunk                 (16 KB)
  __shared__ __bf16 sFT[D * CHUNK];   // phi features TRANSPOSED     (16 KB)
  __shared__ __bf16 sV[CHUNK * D];    // V chunk                     (16 KB)
  __shared__ float  sB[D];

  const size_t xoff = ((size_t)b * TLEN + (size_t)c * CHUNK) * D;
  // warm L2/LDS path for the upcoming chunk loads (global_prefetch_b8)
  __builtin_prefetch(q + xoff + tid * 16, 0, 0);
  __builtin_prefetch(k + xoff + tid * 16, 0, 0);

  for (int i = tid; i < D * D; i += 256) sW[i] = (__bf16)W[i];
  if (tid < D) sB[tid] = bias[tid];
  for (int i = tid; i < CHUNK * D; i += 256) sV[i] = (__bf16)v[xoff + i];

  for (int pass = 0; pass < 2; ++pass) {
    const float* xsrc = (pass == 0) ? q : k;
    __bf16* fdst = ((pass == 0) ? Qf : Kf) + xoff;
    for (int i = tid; i < CHUNK * D; i += 256) sX[i] = (__bf16)xsrc[xoff + i];
    __syncthreads();

    // F[t][e] = phi( sum_d X[t][d]*W[e][d] + b[e] ) : 64x128 out, 32 tiles / 8 waves
    for (int tile = wave; tile < 32; tile += 8) {
      const int row0 = (tile >> 3) * 16, col0 = (tile & 7) * 16;
      f32x8 acc = {};
#pragma unroll
      for (int kk = 0; kk < D; kk += 32) {
        bf16x16 fa = frag_a(sX, row0, kk, D, lane);
        bf16x16 fb = frag_bT(sW, kk, col0, D, lane);  // B[d][e] = W[e][d]
        acc = __builtin_amdgcn_wmma_f32_16x16x32_bf16(false, fa, false, fb,
                                                      (short)0, acc, false, false);
      }
      const int ncol = col0 + (lane & 15);
      const int mbase = row0 + ((lane >> 4) << 3);
#pragma unroll
      for (int r = 0; r < 8; ++r) {
        const int m = mbase + r;
        __bf16 ph = (__bf16)phi_act(acc[r] + sB[ncol]);
        sFT[ncol * CHUNK + m] = ph;      // transposed: [e][t], contiguous gathers later
        fdst[m * D + ncol] = ph;
      }
    }
    __syncthreads();

    if (pass == 1) {
      // chunkKV = Kf^T @ V : [128 x 128], K-dim = 64 ; 64 tiles / 8 waves
      // A[d][t] = sFT[d][t] (row-major, ld = CHUNK) -> coalesced ds_load_b128
      float* ckvp = ckv + (size_t)bc * D * DV;
      for (int tile = wave; tile < 64; tile += 8) {
        const int row0 = (tile >> 3) * 16, col0 = (tile & 7) * 16;
        f32x8 acc = {};
#pragma unroll
        for (int kk = 0; kk < CHUNK; kk += 32) {
          bf16x16 fa = frag_a(sFT, row0, kk, CHUNK, lane);
          bf16x16 fb = frag_b(sV, kk, col0, D, lane);
          acc = __builtin_amdgcn_wmma_f32_16x16x32_bf16(false, fa, false, fb,
                                                        (short)0, acc, false, false);
        }
        const int ncol = col0 + (lane & 15);
        const int mbase = row0 + ((lane >> 4) << 3);
#pragma unroll
        for (int r = 0; r < 8; ++r) ckvp[(mbase + r) * DV + ncol] = acc[r];
      }
      if (tid < D) {  // chunkZ[d] = sum_t Kf[t][d] -- contiguous along sFT row
        float s = 0.f;
        for (int t = 0; t < CHUNK; ++t) s += (float)sFT[tid * CHUNK + t];
        cz[(size_t)bc * D + tid] = s;
      }
    }
    __syncthreads();
  }
}

// =====================================================================
// Pass B: per batch, in-place EXCLUSIVE prefix sum over chunks of
//         chunkKV [NC][D][DV] and chunkZ [NC][D]  (f32, keeps precision)
// =====================================================================
__global__ __launch_bounds__(256) void cla_passB(float* __restrict__ ckv, float* __restrict__ cz) {
  const int b = blockIdx.x;
  const int tid = threadIdx.x;
  float acc[64];
#pragma unroll
  for (int i = 0; i < 64; ++i) acc[i] = 0.f;
  float zacc = 0.f;
  for (int c = 0; c < NC; ++c) {
    float* p = ckv + ((size_t)(b * NC + c)) * D * DV;
#pragma unroll
    for (int i = 0; i < 64; ++i) {
      const int idx = tid + i * 256;
      const float cur = p[idx];
      p[idx] = acc[i];
      acc[i] += cur;
    }
    if (tid < D) {
      float* pz = cz + (size_t)(b * NC + c) * D + tid;
      const float cur = *pz;
      *pz = zacc;
      zacc += cur;
    }
  }
}

// =====================================================================
// Pass C: per (batch, chunk):
//   P   = mask(Qf @ Kf^T)                       (f32 scores, bf16 copy)
//   den = Qf . z_prefix + rowsum(P) + eps       (f32)
//   out = (Qf @ S_prefix + P @ V) / den         (f32 accumulate)
// Qf/Kf staged via global_load_async_to_lds_b128 (ASYNCcnt path).
// =====================================================================
__global__ __launch_bounds__(256) void cla_passC(
    const float* __restrict__ v, const __bf16* __restrict__ Qf, const __bf16* __restrict__ Kf,
    const float* __restrict__ ckv, const float* __restrict__ cz, float* __restrict__ out) {
  const int bc = blockIdx.x;
  const int b = bc / NC, c = bc % NC;
  const int tid = threadIdx.x, lane = tid & 31, wave = tid >> 5;

  __shared__ __bf16 sQ[CHUNK * D];       // 16 KB  (async filled)
  __shared__ __bf16 sK[CHUNK * D];       // 16 KB  (async filled)
  __shared__ __bf16 sV[CHUNK * D];       // 16 KB
  __shared__ __bf16 sS[D * DV];          // 32 KB state prefix (bf16 operands)
  __shared__ __bf16 sP[CHUNK * CHUNK];   // 8 KB
  __shared__ float  sPf[CHUNK * CHUNK];  // 16 KB
  __shared__ float  sDen[CHUNK];
  __shared__ float  sZ[D];

  const size_t xoff = ((size_t)b * TLEN + (size_t)c * CHUNK) * D;

  // --- async bf16 feature tiles: memory -> LDS, no VGPR roundtrip ---
  {
    const void* qbase = (const void*)(Qf + xoff);
    const void* kbase = (const void*)(Kf + xoff);
    const uint32_t sQb = (uint32_t)(uintptr_t)(&sQ[0]);
    const uint32_t sKb = (uint32_t)(uintptr_t)(&sK[0]);
#pragma unroll
    for (int j = 0; j < 4; ++j) {                    // 4 * 256 * 16B = 16 KB each
      const uint32_t off = (uint32_t)(tid + j * 256) * 16u;
      async_copy_b128(sQb + off, qbase, off);
      async_copy_b128(sKb + off, kbase, off);
    }
  }

  // --- regular fills that need f32 -> bf16 conversion (overlap with async) ---
  for (int i = tid; i < CHUNK * D; i += 256) sV[i] = (__bf16)v[xoff + i];
  const float* Sp = ckv + (size_t)bc * D * DV;
  for (int i = tid; i < D * DV; i += 256) sS[i] = (__bf16)Sp[i];
  if (tid < D) sZ[tid] = cz[(size_t)bc * D + tid];

  wait_async0();
  __syncthreads();

  // scores = Qf @ Kf^T : 64x64, 16 tiles / 8 waves
  for (int tile = wave; tile < 16; tile += 8) {
    const int row0 = (tile >> 2) * 16, col0 = (tile & 3) * 16;
    f32x8 acc = {};
#pragma unroll
    for (int kk = 0; kk < D; kk += 32) {
      bf16x16 fa = frag_a(sQ, row0, kk, D, lane);
      bf16x16 fb = frag_bT(sK, kk, col0, D, lane);  // B[d][tau] = Kf[tau][d]
      acc = __builtin_amdgcn_wmma_f32_16x16x32_bf16(false, fa, false, fb,
                                                    (short)0, acc, false, false);
    }
    const int ncol = col0 + (lane & 15);
    const int mbase = row0 + ((lane >> 4) << 3);
#pragma unroll
    for (int r = 0; r < 8; ++r) {
      const int m = mbase + r;
      const float sc = (ncol <= m) ? acc[r] : 0.f;  // causal mask
      sPf[m * CHUNK + ncol] = sc;
      sP[m * CHUNK + ncol] = (__bf16)sc;
    }
  }
  __syncthreads();

  if (tid < CHUNK) {
    float d0 = EPSV;
    for (int e = 0; e < D; ++e) d0 += (float)sQ[tid * D + e] * sZ[e];
    for (int tau = 0; tau < CHUNK; ++tau) d0 += sPf[tid * CHUNK + tau];
    sDen[tid] = d0;
  }
  __syncthreads();

  // out = (Qf @ S + P @ V) / den : 64x128, 32 tiles / 8 waves
  float* op = out + xoff;  // DV == D
  for (int tile = wave; tile < 32; tile += 8) {
    const int row0 = (tile >> 3) * 16, col0 = (tile & 7) * 16;
    f32x8 acc = {};
#pragma unroll
    for (int kk = 0; kk < D; kk += 32) {      // inter-chunk: Qf @ S_prefix
      bf16x16 fa = frag_a(sQ, row0, kk, D, lane);
      bf16x16 fb = frag_b(sS, kk, col0, DV, lane);
      acc = __builtin_amdgcn_wmma_f32_16x16x32_bf16(false, fa, false, fb,
                                                    (short)0, acc, false, false);
    }
#pragma unroll
    for (int kk = 0; kk < CHUNK; kk += 32) {  // intra-chunk: P @ V
      bf16x16 fa = frag_a(sP, row0, kk, CHUNK, lane);
      bf16x16 fb = frag_b(sV, kk, col0, D, lane);
      acc = __builtin_amdgcn_wmma_f32_16x16x32_bf16(false, fa, false, fb,
                                                    (short)0, acc, false, false);
    }
    const int ncol = col0 + (lane & 15);
    const int mbase = row0 + ((lane >> 4) << 3);
#pragma unroll
    for (int r = 0; r < 8; ++r) {
      const int m = mbase + r;
      op[m * DV + ncol] = acc[r] / sDen[m];
    }
  }
}

extern "C" void kernel_launch(void* const* d_in, const int* in_sizes, int n_in,
                              void* d_out, int out_size, void* d_ws, size_t ws_size,
                              hipStream_t stream) {
  (void)in_sizes; (void)n_in; (void)out_size; (void)ws_size;
  const float* q    = (const float*)d_in[0];
  const float* k    = (const float*)d_in[1];
  const float* v    = (const float*)d_in[2];
  const float* W    = (const float*)d_in[3];
  const float* bias = (const float*)d_in[4];
  float* out = (float*)d_out;

  // workspace layout
  char* ws = (char*)d_ws;
  const size_t featBytes = (size_t)BATCH * TLEN * D * sizeof(__bf16);  // 2 MB each
  __bf16* Qf = (__bf16*)ws;
  __bf16* Kf = (__bf16*)(ws + featBytes);
  float*  ckv = (float*)(ws + 2 * featBytes);                          // 8 MB
  float*  cz  = (float*)(ws + 2 * featBytes +
                         (size_t)BATCH * NC * D * DV * sizeof(float)); // 64 KB

  cla_passA<<<BATCH * NC, 256, 0, stream>>>(q, k, v, W, bias, Qf, Kf, ckv, cz);
  cla_passB<<<BATCH, 256, 0, stream>>>(ckv, cz);
  cla_passC<<<BATCH * NC, 256, 0, stream>>>(v, Qf, Kf, ckv, cz, out);
}
